// GCN_3143916060942
// MI455X (gfx1250) — matrix operand order repaired and checked by reading, not compiled, not run on hardware
//
#include <hip/hip_runtime.h>

// ---------------------------------------------------------------------------
// GCN forward on MI455X (gfx1250, wave32, WMMA).
// GEMMs via v_wmma_f32_16x16x32_f16 (f16 in, f32 accum); scatter via
// coalesced float4 gather + global_atomic_add_f32 (bandwidth-bound phase).
// ---------------------------------------------------------------------------

typedef __attribute__((ext_vector_type(16))) _Float16 v16h;
typedef __attribute__((ext_vector_type(8)))  float    v8f;

static constexpr int GN = 100000;   // nodes
static constexpr int GE = 1600000;  // edges
static constexpr int GH = 128;      // hidden
static constexpr int GG = 256;      // graphs
static constexpr int GDEPTH = 3;
static constexpr int GDOUT = 10;

// ---------------------------------------------------------------------------
// Weight pre-swizzle: f32 [128][128] row-major -> f16 fragments.
// Fragment index layout: (((kc*8 + nt)*32 + lane)*16 + t) where
//   lane = hi*16 + nn,  k = kc*32 + 16*hi + t,  n = nt*16 + nn
// so each lane's 16 halves for one B fragment are contiguous (32B load).
// ---------------------------------------------------------------------------
__global__ void convert_weights_kernel(const float* __restrict__ enc_w1,
                                       const float* __restrict__ enc_w2,
                                       const float* __restrict__ conv_w,
                                       const float* __restrict__ dec_w1,
                                       _Float16* __restrict__ wsw) {
  int gid = blockIdx.x * blockDim.x + threadIdx.x;
  if (gid >= 6 * 16384) return;
  int mat  = gid >> 14;
  int idx  = gid & 16383;
  int t    = idx & 15;
  int lane = (idx >> 4) & 31;
  int tile = idx >> 9;          // 0..31 = kc*8 + nt
  int nt = tile & 7, kc = tile >> 3;
  int hi = lane >> 4, nn = lane & 15;
  int k = kc * 32 + 16 * hi + t;
  int n = nt * 16 + nn;
  const float* W;
  switch (mat) {
    case 0: W = enc_w1; break;
    case 1: W = enc_w2; break;
    case 2: W = conv_w; break;
    case 3: W = conv_w + 16384; break;
    case 4: W = conv_w + 32768; break;
    default: W = dec_w1; break;
  }
  wsw[gid] = (_Float16)W[k * 128 + n];
}

// ---------------------------------------------------------------------------
// Elementwise helpers
// ---------------------------------------------------------------------------
__global__ void fill_kernel(float* __restrict__ p, float v, int n) {
  int i = blockIdx.x * blockDim.x + threadIdx.x;
  if (i < n) p[i] = v;
}

__global__ void deg_kernel(const int* __restrict__ dst, float* __restrict__ deg, int ne) {
  int i = blockIdx.x * blockDim.x + threadIdx.x;
  if (i < ne) atomicAdd(&deg[dst[i]], 1.0f);
}

__global__ void rsqrt_kernel(float* __restrict__ p, int n) {
  int i = blockIdx.x * blockDim.x + threadIdx.x;
  if (i < n) p[i] = rsqrtf(p[i]);
}

// h = hw * inv_sqrt^2   (self-loop term; also initializes the scatter output)
__global__ void selfinit_kernel(const float* __restrict__ hw,
                                const float* __restrict__ inv,
                                float* __restrict__ h, int nn) {
  int t = blockIdx.x * blockDim.x + threadIdx.x;   // nn*32 threads
  int node = t >> 5, r = t & 31;
  if (node >= nn) return;
  float iv = inv[node];
  float c = iv * iv;
  const float4 v = *(const float4*)(hw + (size_t)node * GH + r * 4);
  float4 o = make_float4(v.x * c, v.y * c, v.z * c, v.w * c);
  *(float4*)(h + (size_t)node * GH + r * 4) = o;
}

// h[dst] += hw[src] * (inv[src]*inv[dst]) — one wave per edge: 512B coalesced
// gather + 128 contiguous f32 atomics per edge.
__global__ void scatter_kernel(const float* __restrict__ hw,
                               const int* __restrict__ src,
                               const int* __restrict__ dst,
                               const float* __restrict__ inv,
                               float* __restrict__ h, int ne) {
  int t = blockIdx.x * blockDim.x + threadIdx.x;   // ne*32 threads
  int e = t >> 5, r = t & 31;
  if (e >= ne) return;
  int s = src[e], d = dst[e];
  float c = inv[s] * inv[d];
  const float4 v = *(const float4*)(hw + (size_t)s * GH + r * 4);
  float* o = h + (size_t)d * GH + r * 4;
  atomicAdd(o + 0, v.x * c);
  atomicAdd(o + 1, v.y * c);
  atomicAdd(o + 2, v.z * c);
  atomicAdd(o + 3, v.w * c);
}

__global__ void pool_kernel(const float* __restrict__ h,
                            const int* __restrict__ batch,
                            float* __restrict__ pooled, int nn) {
  int t = blockIdx.x * blockDim.x + threadIdx.x;   // nn*32 threads
  int node = t >> 5, r = t & 31;
  if (node >= nn) return;
  int g = batch[node];
  const float4 v = *(const float4*)(h + (size_t)node * GH + r * 4);
  float* o = pooled + (size_t)g * GH + r * 4;
  atomicAdd(o + 0, v.x);
  atomicAdd(o + 1, v.y);
  atomicAdd(o + 2, v.z);
  atomicAdd(o + 3, v.w);
}

// ---------------------------------------------------------------------------
// A-fragment load: 16x32 f16 tile from f32 row-major [*, 128] source.
// Lane l: m = l&15, hi = l>>4; halves j -> K = kc*32 + 16*hi + j.
// Four contiguous float4 loads.
// Optional fused ReLU (defers reference's post-scatter ReLU).
// ---------------------------------------------------------------------------
__device__ __forceinline__ float4 relu4(float4 v) {
  return make_float4(fmaxf(v.x, 0.f), fmaxf(v.y, 0.f), fmaxf(v.z, 0.f), fmaxf(v.w, 0.f));
}

__device__ __forceinline__ v16h load_a_tile(const float* __restrict__ A, int row0,
                                            int kc, int lane, int relu_in) {
  int m = lane & 15, hi = lane >> 4;
  const float* p = A + (size_t)(row0 + m) * GH + kc * 32 + hi * 16;
  float4 f0 = *(const float4*)(p);
  float4 f1 = *(const float4*)(p + 4);
  float4 f2 = *(const float4*)(p + 8);
  float4 f3 = *(const float4*)(p + 12);
  if (relu_in) { f0 = relu4(f0); f1 = relu4(f1); f2 = relu4(f2); f3 = relu4(f3); }
  v16h a;
  a[0]  = (_Float16)f0.x; a[1]  = (_Float16)f0.y; a[2]  = (_Float16)f0.z; a[3]  = (_Float16)f0.w;
  a[4]  = (_Float16)f1.x; a[5]  = (_Float16)f1.y; a[6]  = (_Float16)f1.z; a[7]  = (_Float16)f1.w;
  a[8]  = (_Float16)f2.x; a[9]  = (_Float16)f2.y; a[10] = (_Float16)f2.z; a[11] = (_Float16)f2.w;
  a[12] = (_Float16)f3.x; a[13] = (_Float16)f3.y; a[14] = (_Float16)f3.z; a[15] = (_Float16)f3.w;
  return a;
}

// ---------------------------------------------------------------------------
// out[ntiles*16, 128] = (relu_in? relu(A) : A) @ W + bias
// One wave per 16-row tile; 8 n-tiles; K loop of 4x32; 32 WMMA per tile.
// ---------------------------------------------------------------------------
__global__ void __launch_bounds__(256)
linear_kernel(const float* __restrict__ A, const _Float16* __restrict__ Wsw,
              const float* __restrict__ bias, float* __restrict__ out,
              int ntiles, int relu_in) {
  int wave = blockIdx.x * (blockDim.x >> 5) + (threadIdx.x >> 5);
  if (wave >= ntiles) return;               // wave-uniform: EXEC stays all-ones
  int lane = threadIdx.x & 31;
  int n = lane & 15, hi = lane >> 4;
  int row0 = wave * 16;

  v8f acc[8];
  const v8f zero = {0.f, 0.f, 0.f, 0.f, 0.f, 0.f, 0.f, 0.f};
#pragma unroll
  for (int nt = 0; nt < 8; ++nt) acc[nt] = zero;

  const v16h* wb = (const v16h*)Wsw;
#pragma unroll
  for (int kc = 0; kc < 4; ++kc) {
    v16h a = load_a_tile(A, row0, kc, lane, relu_in);
#pragma unroll
    for (int nt = 0; nt < 8; ++nt) {
      v16h b = wb[(kc * 8 + nt) * 32 + lane];   // 32B contiguous per lane
      acc[nt] = __builtin_amdgcn_wmma_f32_16x16x32_f16(
          false, a, false, b, (short)0, acc[nt], false, false);
    }
  }

#pragma unroll
  for (int nt = 0; nt < 8; ++nt) {
    float bv = bias[nt * 16 + n];
#pragma unroll
    for (int r = 0; r < 8; ++r) {
      out[(size_t)(row0 + 8 * hi + r) * GH + nt * 16 + n] = acc[nt][r] + bv;
    }
  }
}

// ---------------------------------------------------------------------------
// Final decoder: out[256,10] = relu(A[256,128]) @ w2[128,10] + b2
// One wave per 16-row tile; single padded n-tile (cols 10..15 zero).
// ---------------------------------------------------------------------------
__global__ void __launch_bounds__(256)
decoder_out_kernel(const float* __restrict__ A, const float* __restrict__ w2,
                   const float* __restrict__ b2, float* __restrict__ out) {
  int wave = blockIdx.x * (blockDim.x >> 5) + (threadIdx.x >> 5);
  if (wave >= GG / 16) return;
  int lane = threadIdx.x & 31;
  int n = lane & 15, hi = lane >> 4;
  int row0 = wave * 16;

  v8f acc = {0.f, 0.f, 0.f, 0.f, 0.f, 0.f, 0.f, 0.f};
#pragma unroll
  for (int kc = 0; kc < 4; ++kc) {
    v16h a = load_a_tile(A, row0, kc, lane, /*relu_in=*/1);
    v16h b;
#pragma unroll
    for (int t = 0; t < 16; ++t) {
      int k = kc * 32 + 16 * hi + t;
      b[t] = (n < GDOUT) ? (_Float16)w2[k * GDOUT + n] : (_Float16)0.0f;
    }
    acc = __builtin_amdgcn_wmma_f32_16x16x32_f16(
        false, a, false, b, (short)0, acc, false, false);
  }

  if (n < GDOUT) {
    float bv = b2[n];
#pragma unroll
    for (int r = 0; r < 8; ++r) {
      out[(size_t)(row0 + 8 * hi + r) * GDOUT + n] = acc[r] + bv;
    }
  }
}

// ---------------------------------------------------------------------------
// Orchestration
// ---------------------------------------------------------------------------
extern "C" void kernel_launch(void* const* d_in, const int* in_sizes, int n_in,
                              void* d_out, int out_size, void* d_ws, size_t ws_size,
                              hipStream_t stream) {
  const float* x      = (const float*)d_in[0];
  const int*   ei     = (const int*)d_in[1];
  // d_in[2]: control_edge_index (unused by reference)
  const int*   batch  = (const int*)d_in[3];
  const float* enc_w1 = (const float*)d_in[4];
  const float* enc_b1 = (const float*)d_in[5];
  const float* enc_w2 = (const float*)d_in[6];
  const float* enc_b2 = (const float*)d_in[7];
  const float* conv_w = (const float*)d_in[8];
  const float* conv_b = (const float*)d_in[9];
  const float* dec_w1 = (const float*)d_in[10];
  const float* dec_b1 = (const float*)d_in[11];
  const float* dec_w2 = (const float*)d_in[12];
  const float* dec_b2 = (const float*)d_in[13];
  float* out = (float*)d_out;

  const int* src = ei;
  const int* dst = ei + GE;

  // workspace carve (≈103 MB)
  char* w = (char*)d_ws;
  float*    h      = (float*)w;    w += (size_t)GN * GH * 4;
  float*    hw     = (float*)w;    w += (size_t)GN * GH * 4;
  float*    inv    = (float*)w;    w += (size_t)GN * 4;
  float*    pooled = (float*)w;    w += (size_t)GG * GH * 4;
  float*    dbuf   = (float*)w;    w += (size_t)GG * GH * 4;
  _Float16* wsw    = (_Float16*)w; w += (size_t)6 * 16384 * 2;

  // 1) pre-swizzle all 128x128 weights to f16 fragments
  convert_weights_kernel<<<(6 * 16384 + 255) / 256, 256, 0, stream>>>(
      enc_w1, enc_w2, conv_w, dec_w1, wsw);

  // 2) degrees with self-loops -> inv_sqrt
  fill_kernel<<<(GN + 255) / 256, 256, 0, stream>>>(inv, 1.0f, GN);
  deg_kernel<<<(GE + 255) / 256, 256, 0, stream>>>(dst, inv, GE);
  rsqrt_kernel<<<(GN + 255) / 256, 256, 0, stream>>>(inv, GN);

  // 3) encoder MLP: hw = x@W1+b1 ; h = relu(hw)@W2+b2
  const int ntilesN = GN / 16;                 // 6250
  const int blocksN = (ntilesN + 7) / 8;       // 8 waves / block
  linear_kernel<<<blocksN, 256, 0, stream>>>(x,  wsw + 0 * 16384, enc_b1, hw, ntilesN, 0);
  linear_kernel<<<blocksN, 256, 0, stream>>>(hw, wsw + 1 * 16384, enc_b2, h,  ntilesN, 1);

  // 4) GCN layers: hw = relu?(h)@Wc+bc ; h = hw*inv^2 (self-loop) ; scatter
  for (int l = 0; l < GDEPTH; ++l) {
    linear_kernel<<<blocksN, 256, 0, stream>>>(
        h, wsw + (2 + l) * 16384, conv_b + l * GH, hw, ntilesN, (l > 0) ? 1 : 0);
    selfinit_kernel<<<(GN * 32) / 256, 256, 0, stream>>>(hw, inv, h, GN);
    scatter_kernel<<<(GE * 32) / 256, 256, 0, stream>>>(hw, src, dst, inv, h, GE);
  }

  // 5) global add pool (note: last conv layer has no ReLU, pool raw h)
  fill_kernel<<<(GG * GH + 255) / 256, 256, 0, stream>>>(pooled, 0.0f, GG * GH);
  pool_kernel<<<(GN * 32) / 256, 256, 0, stream>>>(h, batch, pooled, GN);

  // 6) decoder: dbuf = pooled@dw1+db1 ; out = relu(dbuf)@dw2+db2
  const int ntilesG = GG / 16;                 // 16
  linear_kernel<<<(ntilesG + 7) / 8, 256, 0, stream>>>(
      pooled, wsw + 5 * 16384, dec_b1, dbuf, ntilesG, 0);
  decoder_out_kernel<<<(ntilesG + 7) / 8, 256, 0, stream>>>(dbuf, dec_w2, dec_b2, out);
}
